// FAEncoder_13932873908934
// MI455X (gfx1250) — matrix operand order repaired and checked by reading, not compiled, not run on hardware
//
#include <hip/hip_runtime.h>
#include <hip/hip_bf16.h>
#include <math.h>

// Problem constants (from reference)
#define HID   256
#define SEQL  512          // N in reference
#define BATCH 8
#define BT    64           // BATCH * 8 sign-flip replicas
#define MROWS (BT * SEQL)  // 32768 GEMM rows

typedef __attribute__((ext_vector_type(16))) __bf16 v16bf;
typedef __attribute__((ext_vector_type(8)))  float  v8f;
typedef __attribute__((ext_vector_type(4)))  unsigned int uint4v;

__device__ __forceinline__ unsigned short f2bf(float x) {
    union { float f; unsigned u; } v; v.f = x;
    unsigned r = v.u + 0x7FFFu + ((v.u >> 16) & 1u);   // round-to-nearest-even
    return (unsigned short)(r >> 16);
}
__device__ __forceinline__ float sigm(float x) { return 1.0f / (1.0f + __expf(-x)); }

// ---------------------------------------------------------------------------
// Repack weight W (K x N, f32, row-major) into bf16 WMMA-B fragment order:
// element o = ((ntile*(K/32) + kt) * 1024) + lane*16 + j
//   holds  W[kt*32 + (lane>>4)*16 + j][ntile*16 + (lane&15)]
// so a GEMM wave fetches a whole B fragment as two contiguous 16B
// global_load_b128's per lane (L2-resident: weights are < 1 MB/layer).
// ---------------------------------------------------------------------------
__global__ void repack_b_kernel(const float* __restrict__ W,
                                unsigned short* __restrict__ Brp,
                                int K, int N) {
    const int o = blockIdx.x * blockDim.x + threadIdx.x;
    if (o >= K * N) return;
    const int block  = o >> 10;
    const int within = o & 1023;
    const int lane   = within >> 4;
    const int j      = within & 15;
    const int ktiles = K >> 5;
    const int ntile  = block / ktiles;
    const int kt     = block - ntile * ktiles;
    const int row    = (kt << 5) + ((lane >> 4) << 4) + j;
    const int col    = (ntile << 4) + (lane & 15);
    Brp[o] = f2bf(W[(size_t)row * N + col]);
}

// ---------------------------------------------------------------------------
// Preprocess: mask, center, 3x3 covariance, Jacobi eigh, projection,
// build h0 (BT x SEQL x 128) in bf16 and mask_pad. One block per batch.
// ---------------------------------------------------------------------------
__global__ void __launch_bounds__(256) prep_kernel(
    const float* __restrict__ X,       // (8, 512, 3)
    const float* __restrict__ hS,      // (8, 512, 125)
    unsigned short* __restrict__ h0b,  // (64, 512, 128) bf16
    unsigned char*  __restrict__ maskpad) // (64, 512)
{
    const int b   = blockIdx.x;
    const int tid = threadIdx.x;
    __shared__ float red4[256][4];
    __shared__ float red6[256][6];
    __shared__ float s_center[3];
    __shared__ float s_V[9];   // s_V[j*3+i] = V[row j][eigvec col i], ascending eigvals

    const float* Xb = X + (size_t)b * SEQL * 3;

    // pass 1: masked mean
    float sm = 0.f, s0 = 0.f, s1 = 0.f, s2 = 0.f;
    for (int p = tid; p < SEQL; p += 256) {
        float x0 = Xb[p*3+0], x1 = Xb[p*3+1], x2 = Xb[p*3+2];
        float m = ((x0 + x1 + x2) != 0.0f) ? 1.0f : 0.0f;
        sm += m; s0 += x0*m; s1 += x1*m; s2 += x2*m;
    }
    red4[tid][0]=sm; red4[tid][1]=s0; red4[tid][2]=s1; red4[tid][3]=s2;
    __syncthreads();
    if (tid == 0) {
        float a0=0,a1=0,a2=0,a3=0;
        for (int t = 0; t < 256; ++t) { a0+=red4[t][0]; a1+=red4[t][1]; a2+=red4[t][2]; a3+=red4[t][3]; }
        float inv = (a0 != 0.f) ? (1.0f / a0) : 0.0f;
        s_center[0]=a1*inv; s_center[1]=a2*inv; s_center[2]=a3*inv;
    }
    __syncthreads();
    const float c0 = s_center[0], c1 = s_center[1], c2 = s_center[2];

    // pass 2: covariance C = Xc^T Xc
    float m00=0,m01=0,m02=0,m11=0,m12=0,m22=0;
    for (int p = tid; p < SEQL; p += 256) {
        float x0 = Xb[p*3+0], x1 = Xb[p*3+1], x2 = Xb[p*3+2];
        float m = ((x0 + x1 + x2) != 0.0f) ? 1.0f : 0.0f;
        float y0 = x0 - c0*m, y1 = x1 - c1*m, y2 = x2 - c2*m;
        m00 += y0*y0; m01 += y0*y1; m02 += y0*y2;
        m11 += y1*y1; m12 += y1*y2; m22 += y2*y2;
    }
    red6[tid][0]=m00; red6[tid][1]=m01; red6[tid][2]=m02;
    red6[tid][3]=m11; red6[tid][4]=m12; red6[tid][5]=m22;
    __syncthreads();
    if (tid == 0) {
        float s[6] = {0,0,0,0,0,0};
        for (int t = 0; t < 256; ++t)
            for (int q = 0; q < 6; ++q) s[q] += red6[t][q];
        float a[3][3] = {{s[0],s[1],s[2]},{s[1],s[3],s[4]},{s[2],s[4],s[5]}};
        float V[3][3] = {{1,0,0},{0,1,0},{0,0,1}};
        for (int it = 0; it < 16; ++it) {           // cyclic Jacobi sweeps
            const int PQ[3][2] = {{0,1},{0,2},{1,2}};
            for (int r = 0; r < 3; ++r) {
                int p = PQ[r][0], q = PQ[r][1];
                float apq = a[p][q];
                if (fabsf(apq) > 1e-12f) {
                    float th = (a[q][q] - a[p][p]) / (2.0f * apq);
                    float t  = ((th >= 0.f) ? 1.f : -1.f) /
                               (fabsf(th) + sqrtf(th*th + 1.0f));
                    float cc = 1.0f / sqrtf(t*t + 1.0f);
                    float ss = t * cc;
                    for (int k = 0; k < 3; ++k) {
                        float akp = a[k][p], akq = a[k][q];
                        a[k][p] = cc*akp - ss*akq;
                        a[k][q] = ss*akp + cc*akq;
                    }
                    for (int k = 0; k < 3; ++k) {
                        float apk = a[p][k], aqk = a[q][k];
                        a[p][k] = cc*apk - ss*aqk;
                        a[q][k] = ss*apk + cc*aqk;
                    }
                    for (int k = 0; k < 3; ++k) {
                        float vkp = V[k][p], vkq = V[k][q];
                        V[k][p] = cc*vkp - ss*vkq;
                        V[k][q] = ss*vkp + cc*vkq;
                    }
                }
            }
        }
        float e[3] = {a[0][0], a[1][1], a[2][2]};
        int ord[3] = {0,1,2};
        for (int i = 0; i < 2; ++i)
            for (int j = 0; j < 2 - i; ++j)
                if (e[ord[j]] > e[ord[j+1]]) { int tmp = ord[j]; ord[j] = ord[j+1]; ord[j+1] = tmp; }
        for (int j = 0; j < 3; ++j)
            for (int i = 0; i < 3; ++i)
                s_V[j*3 + i] = V[j][ord[i]];
    }
    __syncthreads();

    // pass 3: project + write 8 sign-flipped replicas with h_S concat
    const float* hSb = hS + (size_t)b * SEQL * 125;
    for (int p = tid; p < SEQL; p += 256) {
        float x0 = Xb[p*3+0], x1 = Xb[p*3+1], x2 = Xb[p*3+2];
        float m = ((x0 + x1 + x2) != 0.0f) ? 1.0f : 0.0f;
        float y0 = x0 - c0*m, y1 = x1 - c1*m, y2 = x2 - c2*m;
        float pr[3];
        for (int i = 0; i < 3; ++i)
            pr[i] = y0 * s_V[0*3+i] + y1 * s_V[1*3+i] + y2 * s_V[2*3+i];
        unsigned char pad = (m != 0.0f) ? 0 : 1;
        for (int o = 0; o < 8; ++o) {
            float sg0 = (o & 4) ? 1.f : -1.f;
            float sg1 = (o & 2) ? 1.f : -1.f;
            float sg2 = (o & 1) ? 1.f : -1.f;
            size_t base = ((size_t)(b*8 + o) * SEQL + p) * 128;
            h0b[base + 0] = f2bf(pr[0] * sg0);
            h0b[base + 1] = f2bf(pr[1] * sg1);
            h0b[base + 2] = f2bf(pr[2] * sg2);
            for (int ch = 0; ch < 125; ++ch)
                h0b[base + 3 + ch] = f2bf(hSb[p*125 + ch]);
            maskpad[(size_t)(b*8 + o) * SEQL + p] = pad;
        }
    }
}

// ---------------------------------------------------------------------------
// WMMA bf16 GEMM: C(MxN) = A(MxK) * Brp(K x N, fragment-repacked), row-major C.
// 256 threads = 8 wave32; block tile 128(M) x 128(N); wave tile 32 x 64 =
// 2x4 v_wmma_f32_16x16x32_bf16. A is double-buffered in LDS (one barrier per
// K-step, next tile's global loads issued before the barrier); B fragments are
// loaded straight from L2-resident repacked weights (2x global_load_b128/lane).
// ---------------------------------------------------------------------------
template<bool OUT_BF16>
__global__ void __launch_bounds__(256) wmma_gemm_kernel(
    const unsigned short* __restrict__ A,    // M x K bf16 row-major
    const unsigned short* __restrict__ Brp,  // repacked B fragments
    void* __restrict__ Cout,
    int M, int N, int K)
{
    __shared__ __align__(16) unsigned short ldsA[2][128 * 40]; // [row][k] pitch 40

    const int tid   = threadIdx.x;
    const int lane  = tid & 31;
    const int wid   = tid >> 5;
    const int waveM = wid & 3;   // 0..3 -> M offset *32
    const int waveN = wid >> 2;  // 0..1 -> N offset *64
    const int half  = (lane >> 4) & 1;
    const int l16   = lane & 15;

    const int rowBase = blockIdx.y * 128;
    const int colBase = blockIdx.x * 128;
    const int ktiles  = K >> 5;

    // Per-thread A staging: 2 x 16B vectors covering 128 rows x 32 k
    const int ar  = tid >> 2;          // 0..63
    const int ac8 = (tid & 3) << 3;    // 0,8,16,24
    const unsigned short* pA0 = A + (size_t)(rowBase + ar) * K + ac8;
    const unsigned short* pA1 = A + (size_t)(rowBase + ar + 64) * K + ac8;
    const int lo0 = ar * 40 + ac8;
    const int lo1 = (ar + 64) * 40 + ac8;

    // B fragment base for this wave (n-tile index = blockIdx.x*8 + waveN*4 + nt)
    const unsigned short* pB =
        Brp + ((size_t)(blockIdx.x * 8 + waveN * 4) * ktiles) * 1024 + lane * 16;

    v8f acc[2][4];
    for (int i = 0; i < 2; ++i)
        for (int j = 0; j < 4; ++j)
            for (int r = 0; r < 8; ++r) acc[i][j][r] = 0.0f;

    union Frag { uint4v q[2]; v16bf v; };

    // Prologue: stage k-tile 0 into buffer 0
    {
        uint4v a0 = *(const uint4v*)(pA0);
        uint4v a1 = *(const uint4v*)(pA1);
        *(uint4v*)(&ldsA[0][lo0]) = a0;
        *(uint4v*)(&ldsA[0][lo1]) = a1;
    }

    for (int kt = 0; kt < ktiles; ++kt) {
        const int cur = kt & 1;

        // Issue next A tile's global loads early (consumed after the WMMAs)
        uint4v na0, na1;
        const bool more = (kt + 1 < ktiles);
        if (more) {
            na0 = *(const uint4v*)(pA0 + ((kt + 1) << 5));
            na1 = *(const uint4v*)(pA1 + ((kt + 1) << 5));
        }
        // B fragments for this k-tile: straight from global (L2 hit)
        Frag bF[4];
        #pragma unroll
        for (int nt = 0; nt < 4; ++nt) {
            const unsigned short* bp = pB + ((size_t)nt * ktiles + kt) * 1024;
            bF[nt].q[0] = *(const uint4v*)(bp);
            bF[nt].q[1] = *(const uint4v*)(bp + 8);
        }

        __syncthreads();   // buffer `cur` staged; buffer `cur^1` free to overwrite

        // A fragments (contiguous K layout matching repacked B: elems 0..15 = K half*16..)
        Frag aF[2];
        #pragma unroll
        for (int mt = 0; mt < 2; ++mt) {
            const int row = waveM * 32 + mt * 16 + l16;
            aF[mt].q[0] = *(const uint4v*)(&ldsA[cur][row * 40 + half * 16]);
            aF[mt].q[1] = *(const uint4v*)(&ldsA[cur][row * 40 + half * 16 + 8]);
        }

        #pragma unroll
        for (int mt = 0; mt < 2; ++mt)
            #pragma unroll
            for (int nt = 0; nt < 4; ++nt)
                acc[mt][nt] = __builtin_amdgcn_wmma_f32_16x16x32_bf16(
                    false, aF[mt].v, false, bF[nt].v,
                    (short)0, acc[mt][nt], false, false);

        // Stage next tile into the other buffer; prefetch the one after
        if (more) {
            *(uint4v*)(&ldsA[cur ^ 1][lo0]) = na0;
            *(uint4v*)(&ldsA[cur ^ 1][lo1]) = na1;
            if (kt + 2 < ktiles)
                __builtin_prefetch(pA0 + ((kt + 2) << 5), 0, 1);
        }
    }

    // Store: C/D layout VGPR r -> M = r + 8*(lane>=16), N = lane%16
    #pragma unroll
    for (int mt = 0; mt < 2; ++mt)
        #pragma unroll
        for (int nt = 0; nt < 4; ++nt) {
            const int row0 = rowBase + waveM * 32 + mt * 16 + half * 8;
            const int col  = colBase + waveN * 64 + nt * 16 + l16;
            #pragma unroll
            for (int r = 0; r < 8; ++r) {
                const size_t idx = (size_t)(row0 + r) * N + col;
                if (OUT_BF16) ((unsigned short*)Cout)[idx] = f2bf(acc[mt][nt][r]);
                else          ((float*)Cout)[idx]          = acc[mt][nt][r];
            }
        }
}

// ---------------------------------------------------------------------------
// Bidirectional SRU recurrence. One thread per (row b, direction d, hidden j):
// 64*2*256 = 32768 independent chains of length 512, coalesced along j.
// For kdim==3 h/resSrc alias (in-place): each element is read then overwritten
// by the same owning thread, so no cross-thread hazard.
// ---------------------------------------------------------------------------
__global__ void __launch_bounds__(256) sru_scan_kernel(
    const float* __restrict__ U,
    const float* __restrict__ resSrc,
    int kdim,
    const float* __restrict__ wc,     // (2, 512)
    const float* __restrict__ bias,   // (2, 512)
    const unsigned char* __restrict__ maskpad,
    float* __restrict__ h,            // (64, 512, 512)
    unsigned short* __restrict__ hb)  // bf16 copy for next layer GEMM
{
    const int idx = blockIdx.x * blockDim.x + threadIdx.x;
    const int b = idx >> 9;
    const int d = (idx >> 8) & 1;
    const int j = idx & 255;
    const int ch = d * 256 + j;
    const float vf  = wc[ch],        vr  = wc[512 + ch];
    const float bfv = bias[ch],      brv = bias[512 + ch];
    const size_t kh = (size_t)kdim * 256;

    float c = 0.0f;
    for (int t = 0; t < SEQL; ++t) {
        const int l = d ? (SEQL - 1 - t) : t;
        const size_t urow = ((size_t)(b * SEQL + l) * 2 + d) * kh + j;
        const float u0 = U[urow];
        const float u1 = U[urow + 256];
        const float u2 = U[urow + 512];
        const float rin = (kdim == 4) ? U[urow + 768]
                                      : resSrc[(size_t)(b * SEQL + l) * 512 + ch];
        float hv = 0.0f;
        if (!maskpad[b * SEQL + l]) {
            const float f  = sigm(u1 + vf * c + bfv);
            const float cn = f * c + (1.0f - f) * u0;
            const float r  = sigm(u2 + vr * cn + brv);
            hv = r * cn + (1.0f - r) * rin;
            c = cn;
        }
        const size_t oidx = (size_t)(b * SEQL + l) * 512 + ch;
        h[oidx]  = hv;
        hb[oidx] = f2bf(hv);
    }
}

// ---------------------------------------------------------------------------
// Mean over the 8 replicas: out (8,512,512)
// ---------------------------------------------------------------------------
__global__ void mean_kernel(const float* __restrict__ h, float* __restrict__ out, int n) {
    int i = blockIdx.x * blockDim.x + threadIdx.x;
    if (i >= n) return;
    const int b = i / (SEQL * 512);
    const int r = i % (SEQL * 512);
    float s = 0.0f;
    #pragma unroll
    for (int o = 0; o < 8; ++o)
        s += h[(size_t)(b * 8 + o) * (SEQL * 512) + r];
    out[i] = s * 0.125f;
}

// ---------------------------------------------------------------------------
extern "C" void kernel_launch(void* const* d_in, const int* in_sizes, int n_in,
                              void* d_out, int out_size, void* d_ws, size_t ws_size,
                              hipStream_t stream)
{
    (void)in_sizes; (void)n_in; (void)ws_size;
    const float* X  = (const float*)d_in[0];
    const float* hS = (const float*)d_in[1];
    const float* wp[3] = {(const float*)d_in[2], (const float*)d_in[6],  (const float*)d_in[10]};
    const float* w [3] = {(const float*)d_in[3], (const float*)d_in[7],  (const float*)d_in[11]};
    const float* wc[3] = {(const float*)d_in[4], (const float*)d_in[8],  (const float*)d_in[12]};
    const float* bs[3] = {(const float*)d_in[5], (const float*)d_in[9],  (const float*)d_in[13]};

    // Workspace layout (≈ 400 MB total; assumes ws_size is sized for the problem)
    char* ws = (char*)d_ws;
    size_t off = 0;
    auto alloc = [&](size_t bytes) -> void* {
        void* p = ws + off;
        off = (off + bytes + 255) & ~(size_t)255;
        return p;
    };
    unsigned short* h0b  = (unsigned short*)alloc((size_t)MROWS * 128 * 2);
    unsigned short* Tb   = (unsigned short*)alloc((size_t)MROWS * 256 * 2);
    float*          Uf   = (float*)         alloc((size_t)MROWS * 2048 * 4);
    float*          hF   = (float*)         alloc((size_t)MROWS * 512 * 4);
    unsigned short* hB   = (unsigned short*)alloc((size_t)MROWS * 512 * 2);
    unsigned char*  mpad = (unsigned char*) alloc((size_t)MROWS);
    const int wpK[3] = {128, 512, 512}, wpNn = 256;
    const int wKk    = 256;
    const int wNn[3] = {2048, 1536, 1536};
    unsigned short* wpb[3];
    unsigned short* wb[3];
    for (int l = 0; l < 3; ++l) {
        wpb[l] = (unsigned short*)alloc((size_t)wpK[l] * wpNn * 2);
        wb [l] = (unsigned short*)alloc((size_t)wKk * wNn[l] * 2);
    }

    // Repack weights into bf16 WMMA fragment order (once; L2-resident after)
    for (int l = 0; l < 3; ++l) {
        int n1 = wpK[l] * wpNn;
        int n2 = wKk * wNn[l];
        repack_b_kernel<<<(n1 + 255) / 256, 256, 0, stream>>>(wp[l], wpb[l], wpK[l], wpNn);
        repack_b_kernel<<<(n2 + 255) / 256, 256, 0, stream>>>(w[l],  wb[l],  wKk,    wNn[l]);
    }

    // PCA preprocess + replica expansion
    prep_kernel<<<BATCH, 256, 0, stream>>>(X, hS, h0b, mpad);

    const unsigned short* xin = h0b;
    int din = 128;
    for (int l = 0; l < 3; ++l) {
        const int kmat = (l == 0) ? 4 : 3;
        const int n2   = 2 * HID * kmat;     // 2048 or 1536
        // GEMM1: T = xin (M x din) @ wp (din x 256) -> bf16
        dim3 g1(256 / 128, MROWS / 128);
        wmma_gemm_kernel<true><<<g1, 256, 0, stream>>>(xin, wpb[l], (void*)Tb,
                                                       MROWS, 256, din);
        // GEMM2: U = T (M x 256) @ w (256 x n2) -> f32
        dim3 g2(n2 / 128, MROWS / 128);
        wmma_gemm_kernel<false><<<g2, 256, 0, stream>>>(Tb, wb[l], (void*)Uf,
                                                        MROWS, n2, 256);
        // Bidirectional recurrence (writes f32 h + bf16 copy for next layer)
        sru_scan_kernel<<<(BT * 2 * HID) / 256, 256, 0, stream>>>(
            Uf, hF, kmat, wc[l], bs[l], mpad, hF, hB);
        xin = hB;
        din = 512;
    }

    // Mean over 8 replicas -> d_out (8, 512, 512) f32
    mean_kernel<<<(out_size + 255) / 256, 256, 0, stream>>>(hF, (float*)d_out, out_size);
}
